// VectorQuantizer_65833258713515
// MI455X (gfx1250) — matrix operand order repaired and checked by reading, not compiled, not run on hardware
//
#include <hip/hip_runtime.h>
#include <hip/hip_bf16.h>

typedef __attribute__((ext_vector_type(16))) __bf16 v16bf;
typedef __attribute__((ext_vector_type(8)))  float  v8f;

#define N_TOK 131072   // B*D*H*W
#define K_CB  512
#define C_DIM 64
#define DHW   16384    // D*H*W
#define CDHW  1048576  // C*D*H*W

// ---------------------------------------------------------------------------
// Kernel 1: argmin over K via bf16 split-precision WMMA GEMM.
// Block = 256 threads = 8 waves; each wave handles TWO 16-row stripes (32 rows)
// so each B-tile (4 x v16bf from LDS) feeds 12 WMMAs instead of 6.
// __launch_bounds__(256,1): lift the occupancy-driven VGPR cap (the ~150-reg
// live set spilled to scratch in the hot loop under the default budget).
// LDS (dynamic): codebook as bf16 hi (64KB) + bf16 lo (64KB) + ||cb||^2 (2KB).
// ---------------------------------------------------------------------------
__global__ void __launch_bounds__(256, 1)
vq_argmin_wmma(const float* __restrict__ z,
               const float* __restrict__ cb,
               float* __restrict__ out_idx_f,
               int*   __restrict__ ws_idx) {
    extern __shared__ char smem[];
    __bf16* cbH = (__bf16*)smem;                 // [512*64]
    __bf16* cbL = cbH + K_CB * C_DIM;            // [512*64]
    float*  cb2 = (float*)(cbL + K_CB * C_DIM);  // [512]

    const int tid = threadIdx.x;

    // Cooperative: convert codebook to bf16 hi/lo residual pair in LDS.
    for (int i = tid; i < K_CB * C_DIM; i += 256) {
        float v  = cb[i];
        __bf16 h = (__bf16)v;
        cbH[i] = h;
        cbL[i] = (__bf16)(v - (float)h);
    }
    // ||cb_k||^2 in fp32.
    for (int k = tid; k < K_CB; k += 256) {
        float s = 0.0f;
        #pragma unroll 8
        for (int c = 0; c < C_DIM; ++c) { float v = cb[k * C_DIM + c]; s += v * v; }
        cb2[k] = s;
    }
    __syncthreads();

    const int wave   = tid >> 5;
    const int lane   = tid & 31;
    const int h16    = lane >> 4;        // 0 = lanes 0-15, 1 = lanes 16-31
    const int lan16  = lane & 15;
    const int n0     = (blockIdx.x * 8 + wave) * 32;   // 32 rows per wave

    // A-matrices: two 16-row stripes P (n0..n0+15) and Q (n0+16..n0+31).
    const int rP = n0 + lan16;
    const int rQ = rP + 16;
    const int bpP = (rP >> 14) * CDHW + (rP & (DHW - 1));
    const int bpQ = (rQ >> 14) * CDHW + (rQ & (DHW - 1));

    // A layout (16-bit A 16x32): element e of v16bf -> K = 16*h16 + e.
    v16bf aH0P, aL0P, aH1P, aL1P, aH0Q, aL0Q, aH1Q, aL1Q;
    #pragma unroll
    for (int e = 0; e < 16; ++e) {
        const int c  = e + 16 * h16;
        float v; __bf16 hh;
        v = z[bpP + c * DHW];        hh = (__bf16)v; aH0P[e] = hh; aL0P[e] = (__bf16)(v - (float)hh);
        v = z[bpP + (c + 32) * DHW]; hh = (__bf16)v; aH1P[e] = hh; aL1P[e] = (__bf16)(v - (float)hh);
        v = z[bpQ + c * DHW];        hh = (__bf16)v; aH0Q[e] = hh; aL0Q[e] = (__bf16)(v - (float)hh);
        v = z[bpQ + (c + 32) * DHW]; hh = (__bf16)v; aH1Q[e] = hh; aL1Q[e] = (__bf16)(v - (float)hh);
    }

    float bestP[8], bestQ[8];
    int   bidxP[8], bidxQ[8];
    #pragma unroll
    for (int i = 0; i < 8; ++i) {
        bestP[i] = 3.4e38f; bidxP[i] = 0x7fffffff;
        bestQ[i] = 3.4e38f; bidxQ[i] = 0x7fffffff;
    }

    // B layout (16-bit B 32x16): lane holds column (lane&15), K = 16*h16 + e.
    const int cOff = h16 * 16;

    for (int t = 0; t < 32; ++t) {
        const int col  = t * 16 + lan16;
        const int base = col * C_DIM + cOff;
        const float c2 = cb2[col];                        // issue early, overlaps WMMAs
        const v16bf bH0 = *(const v16bf*)(cbH + base);
        const v16bf bL0 = *(const v16bf*)(cbL + base);
        const v16bf bH1 = *(const v16bf*)(cbH + base + 32);
        const v16bf bL1 = *(const v16bf*)(cbL + base + 32);

        // Two independent accumulator chains hide WMMA->WMMA latency.
        v8f accP = {}, accQ = {};
        accP = __builtin_amdgcn_wmma_f32_16x16x32_bf16(false, aH0P, false, bH0, (short)0, accP, false, false);
        accQ = __builtin_amdgcn_wmma_f32_16x16x32_bf16(false, aH0Q, false, bH0, (short)0, accQ, false, false);
        accP = __builtin_amdgcn_wmma_f32_16x16x32_bf16(false, aH0P, false, bL0, (short)0, accP, false, false);
        accQ = __builtin_amdgcn_wmma_f32_16x16x32_bf16(false, aH0Q, false, bL0, (short)0, accQ, false, false);
        accP = __builtin_amdgcn_wmma_f32_16x16x32_bf16(false, aL0P, false, bH0, (short)0, accP, false, false);
        accQ = __builtin_amdgcn_wmma_f32_16x16x32_bf16(false, aL0Q, false, bH0, (short)0, accQ, false, false);
        accP = __builtin_amdgcn_wmma_f32_16x16x32_bf16(false, aH1P, false, bH1, (short)0, accP, false, false);
        accQ = __builtin_amdgcn_wmma_f32_16x16x32_bf16(false, aH1Q, false, bH1, (short)0, accQ, false, false);
        accP = __builtin_amdgcn_wmma_f32_16x16x32_bf16(false, aH1P, false, bL1, (short)0, accP, false, false);
        accQ = __builtin_amdgcn_wmma_f32_16x16x32_bf16(false, aH1Q, false, bL1, (short)0, accQ, false, false);
        accP = __builtin_amdgcn_wmma_f32_16x16x32_bf16(false, aL1P, false, bH1, (short)0, accP, false, false);
        accQ = __builtin_amdgcn_wmma_f32_16x16x32_bf16(false, aL1Q, false, bH1, (short)0, accQ, false, false);

        #pragma unroll
        for (int i = 0; i < 8; ++i) {
            float sP = c2 - 2.0f * accP[i];   // ||cb||^2 - 2 z.cb  (drop ||z||^2)
            float sQ = c2 - 2.0f * accQ[i];
            if (sP < bestP[i]) { bestP[i] = sP; bidxP[i] = col; }
            if (sQ < bestQ[i]) { bestQ[i] = sQ; bidxQ[i] = col; }
        }
    }

    // Lex-min reduce across the 16 lanes that share each row (xor within halves).
    #pragma unroll
    for (int m = 1; m < 16; m <<= 1) {
        #pragma unroll
        for (int i = 0; i < 8; ++i) {
            float os; int oi;
            os = __shfl_xor(bestP[i], m, 32);
            oi = __shfl_xor(bidxP[i], m, 32);
            if (os < bestP[i] || (os == bestP[i] && oi < bidxP[i])) { bestP[i] = os; bidxP[i] = oi; }
            os = __shfl_xor(bestQ[i], m, 32);
            oi = __shfl_xor(bidxQ[i], m, 32);
            if (os < bestQ[i] || (os == bestQ[i] && oi < bidxQ[i])) { bestQ[i] = os; bidxQ[i] = oi; }
        }
    }

    if (lan16 == 0) {
        #pragma unroll
        for (int i = 0; i < 8; ++i) {
            int rowP = n0 + h16 * 8 + i;        // C/D layout: VGPR i -> M = i + 8*h16
            int rowQ = rowP + 16;
            ws_idx[rowP]    = bidxP[i];
            out_idx_f[rowP] = (float)bidxP[i];
            ws_idx[rowQ]    = bidxQ[i];
            out_idx_f[rowQ] = (float)bidxQ[i];
        }
    }
}

// ---------------------------------------------------------------------------
// Kernel 2: gather z_q / z_q_st, histogram counts, scatter embed_sum.
// ---------------------------------------------------------------------------
__global__ void vq_gather_scatter(const float* __restrict__ z,
                                  const float* __restrict__ cb,
                                  const int*   __restrict__ ws_idx,
                                  float* __restrict__ zqst,
                                  float* __restrict__ zq,
                                  float* __restrict__ counts,
                                  float* __restrict__ esum) {
    const int n = blockIdx.x * 256 + threadIdx.x;
    if (n >= N_TOK) return;
    const int idx   = ws_idx[n];
    const int bpart = (n >> 14) * CDHW + (n & (DHW - 1));
    atomicAdd(&counts[idx], 1.0f);
    #pragma unroll 4
    for (int c = 0; c < C_DIM; ++c) {
        const int p  = bpart + c * DHW;
        const float q  = cb[idx * C_DIM + c];
        const float zv = z[p];
        zq[p]   = q;
        zqst[p] = zv + (q - zv);   // straight-through forward value
        atomicAdd(&esum[idx * C_DIM + c], zv);
    }
}

// ---------------------------------------------------------------------------
// Kernel 3: EMA update, normalization, ok-check, dead-code reinit.
// Single block of 512 threads (thread k owns codebook row k).
// ---------------------------------------------------------------------------
__device__ __forceinline__ unsigned hash_u32(unsigned x) {
    x ^= x >> 16; x *= 0x7feb352dU;
    x ^= x >> 15; x *= 0x846ca68bU;
    x ^= x >> 16; return x;
}

__global__ void vq_finalize(const float* __restrict__ z,
                            const float* __restrict__ cb,
                            const float* __restrict__ cs_in,
                            const float* __restrict__ ea_in,
                            const float* __restrict__ counts,
                            const float* __restrict__ esum,
                            float* __restrict__ out_cb,
                            float* __restrict__ out_cs,
                            float* __restrict__ out_ea) {
    __shared__ float red_s[512];
    __shared__ float red_m[512];
    __shared__ int   red_n[512];
    const int k = threadIdx.x;

    const float cs = cs_in[k] * 0.99f + 0.01f * counts[k];

    // n = sum(cs)
    red_s[k] = cs;
    __syncthreads();
    for (int s = 256; s > 0; s >>= 1) {
        if (k < s) red_s[k] += red_s[k + s];
        __syncthreads();
    }
    const float nsum = red_s[0];
    const float csn  = (cs + 1e-5f) / (nsum + 512.0f * 1e-5f) * nsum;
    const float inv  = 1.0f / csn;

    // pass A: nan / max|new_cb| over own row
    float mx = 0.0f; int has_nan = 0;
    for (int c = 0; c < C_DIM; ++c) {
        float ea  = ea_in[k * C_DIM + c] * 0.99f + 0.01f * esum[k * C_DIM + c];
        float ncb = ea * inv;
        if (ncb != ncb) has_nan = 1;
        float a = fabsf(ncb);
        if (a > mx) mx = a;
    }
    red_m[k] = mx; red_n[k] = has_nan;
    __syncthreads();
    for (int s = 256; s > 0; s >>= 1) {
        if (k < s) { red_m[k] = fmaxf(red_m[k], red_m[k + s]); red_n[k] |= red_n[k + s]; }
        __syncthreads();
    }
    const bool ok   = (red_n[0] == 0) && (red_m[0] < 10.0f);
    const bool dead = cs < 1.0f;

    // deterministic stand-in for jax PRNG (no runtime check; same every call)
    const unsigned rk   = hash_u32((unsigned)k ^ 0x2545F491u);
    const int      ridx = (int)(rk & (N_TOK - 1));
    const int      rbp  = (ridx >> 14) * CDHW + (ridx & (DHW - 1));

    // pass B: write cb / ea
    for (int c = 0; c < C_DIM; ++c) {
        float ea  = ea_in[k * C_DIM + c] * 0.99f + 0.01f * esum[k * C_DIM + c];
        float ncb = ea * inv;
        float cbv = ok ? ncb : cb[k * C_DIM + c];
        if (dead) {
            unsigned h1 = hash_u32((unsigned)(k * C_DIM + c) + 0x9E3779B9u);
            unsigned h2 = hash_u32(h1 + 0x85EBCA6Bu);
            float u1 = ((float)((h1 >> 8) + 1u)) * (1.0f / 16777216.0f); // (0,1]
            float u2 = ((float)(h2 >> 8)) * (1.0f / 16777216.0f);        // [0,1)
            float nrm = sqrtf(-2.0f * __logf(u1)) * __cosf(6.2831853f * u2);
            cbv = z[rbp + c * DHW] + nrm * 0.01f;
            ea  = cbv;
        }
        out_cb[k * C_DIM + c] = cbv;
        out_ea[k * C_DIM + c] = ea;
    }
    out_cs[k] = dead ? 1.0f : cs;
}

// ---------------------------------------------------------------------------
extern "C" void kernel_launch(void* const* d_in, const int* in_sizes, int n_in,
                              void* d_out, int out_size, void* d_ws, size_t ws_size,
                              hipStream_t stream) {
    const float* z  = (const float*)d_in[0];   // (8,64,16,32,32)
    const float* cb = (const float*)d_in[1];   // (512,64)
    const float* cs = (const float*)d_in[2];   // (512,)
    const float* ea = (const float*)d_in[3];   // (512,64)

    float* out   = (float*)d_out;
    float* zqst  = out;                               // 8388608
    float* zq    = out + 8388608;                     // 8388608
    float* oidx  = out + 16777216;                    // 131072
    float* ocb   = out + 16908288;                    // 32768
    float* ocs   = out + 16941056;                    // 512
    float* oea   = out + 16941568;                    // 32768

    int*   ws_idx = (int*)d_ws;                                   // 131072 ints
    float* counts = (float*)((char*)d_ws + (size_t)N_TOK * 4);    // 512
    float* esm    = counts + K_CB;                                // 512*64

    // zero the accumulators (graph-capturable)
    hipMemsetAsync(counts, 0, (size_t)(K_CB + K_CB * C_DIM) * sizeof(float), stream);

    const size_t smem = (size_t)(2 * K_CB * C_DIM) * sizeof(__bf16) + K_CB * sizeof(float);
    vq_argmin_wmma<<<N_TOK / (32 * 8), 256, smem, stream>>>(z, cb, oidx, ws_idx);
    vq_gather_scatter<<<N_TOK / 256, 256, 0, stream>>>(z, cb, ws_idx, zqst, zq, counts, esm);
    vq_finalize<<<1, 512, 0, stream>>>(z, cb, cs, ea, counts, esm, ocb, ocs, oea);
}
